// MoNet_40699110097228
// MI455X (gfx1250) — compile-verified, matrix operand order checked by reading, not run on hardware
//
#include <hip/hip_runtime.h>
#include <hip/hip_bf16.h>

// ---------------------------------------------------------------------------
// MoNet (GMMConv) forward, CDNA5/gfx1250.
//   - Node-side reformulation: g_k = h @ fc[l,k]^T via f16 WMMA (16x16x32),
//     then per-edge msg = sum_k w_ek * g_k[row_e], scatter-add via
//     global_atomic_add_f32.
//   - All weight matrices pre-converted to f16 once (no in-loop cvt).
//   - g_k stored f16 (halves gather traffic; 38MB fits in 192MB L2).
//   - Edge Gaussian weights recomputed on the fly (cheap VALU).
// ---------------------------------------------------------------------------

typedef _Float16 v8h  __attribute__((ext_vector_type(8)));
typedef _Float16 v16h __attribute__((ext_vector_type(16)));
typedef float    v8f  __attribute__((ext_vector_type(8)));

#define HDIM 128
#define LDS_STRIDE 136   // 128 + 8 halfs pad (272B row stride, stays 16B aligned)

// ---------------------------------------------------------------------------
// GEMM: Y[m,n] = sum_k A16[m,k] * W16[n,k]  (+ bias[n])
// A16: [M,128] f16 row-major, W16: [128,128] f16 row-major (used as B = W^T).
// Block = 128 threads = 4 waves; each wave owns TWO 16x16 column tiles
// (cols 32w..32w+31), reusing its A fragment for 2 WMMAs per K chunk.
// Block covers 16 rows x 128 cols. M must be a multiple of 16 (50000 = 3125*16).
// ---------------------------------------------------------------------------
__global__ void __launch_bounds__(128)
gemm_abt_wmma(const _Float16* __restrict__ A,
              const _Float16* __restrict__ W16,
              const float* __restrict__ bias,   // may be null
              float* __restrict__ Yf,           // may be null
              _Float16* __restrict__ Yh,        // may be null
              int M)
{
    __shared__ _Float16 lds_a[16 * LDS_STRIDE];

    const int t    = threadIdx.x;
    const int wave = t >> 5;          // 0..3 -> column tile pair
    const int lane = t & 31;
    const long rowBase = (long)blockIdx.x * 16;

    // Stage the 16x128 f16 A tile into LDS: 128 threads x 16 halfs each.
    {
        const int r  = t >> 3;         // 0..15
        const int cb = (t & 7) * 16;   // 0..112
        const _Float16* ap = A + (rowBase + r) * HDIM + cb;
        v8h v0 = *(const v8h*)(ap);
        v8h v1 = *(const v8h*)(ap + 8);
        *(v8h*)(&lds_a[r * LDS_STRIDE + cb])     = v0;
        *(v8h*)(&lds_a[r * LDS_STRIDE + cb + 8]) = v1;
    }
    __syncthreads();

    const int col0 = wave * 32 + (lane & 15);   // first tile's output column
    const int col1 = col0 + 16;                 // second tile's output column
    const int kb   = (lane >> 4) * 8;           // 16-bit A/B fragment K sub-offset
    const int arow = lane & 15;

    v8f c0 = {0.f, 0.f, 0.f, 0.f, 0.f, 0.f, 0.f, 0.f};
    v8f c1 = {0.f, 0.f, 0.f, 0.f, 0.f, 0.f, 0.f, 0.f};

#pragma unroll
    for (int kk = 0; kk < HDIM; kk += 32) {
        // A fragment (16x32): elems 0..7 = K kk+kb..+7, elems 8..15 = K kk+16+kb..+7
        v8h alo = *(const v8h*)(&lds_a[arow * LDS_STRIDE + kk + kb]);
        v8h ahi = *(const v8h*)(&lds_a[arow * LDS_STRIDE + kk + 16 + kb]);
        v16h a;
#pragma unroll
        for (int i = 0; i < 8; ++i) { a[i] = alo[i]; a[8 + i] = ahi[i]; }

        // Two B fragments (32x16): B[k, col] = W16[col, k], pure f16 loads.
        const _Float16* wp0 = W16 + (long)col0 * HDIM + kk + kb;
        const _Float16* wp1 = W16 + (long)col1 * HDIM + kk + kb;
        v8h b0lo = *(const v8h*)(wp0);
        v8h b0hi = *(const v8h*)(wp0 + 16);
        v8h b1lo = *(const v8h*)(wp1);
        v8h b1hi = *(const v8h*)(wp1 + 16);
        v16h b0, b1;
#pragma unroll
        for (int i = 0; i < 8; ++i) {
            b0[i] = b0lo[i]; b0[8 + i] = b0hi[i];
            b1[i] = b1lo[i]; b1[8 + i] = b1hi[i];
        }

        c0 = __builtin_amdgcn_wmma_f32_16x16x32_f16(false, a, false, b0,
                                                    (short)0, c0, false, false);
        c1 = __builtin_amdgcn_wmma_f32_16x16x32_f16(false, a, false, b1,
                                                    (short)0, c1, false, false);
    }

    const float bv0 = bias ? bias[col0] : 0.f;
    const float bv1 = bias ? bias[col1] : 0.f;
#pragma unroll
    for (int j = 0; j < 8; ++j) {
        const long r = rowBase + j + 8 * (lane >> 4);   // D layout: lane>=16 -> M+8
        if (r < M) {
            const float v0 = c0[j] + bv0;
            const float v1 = c1[j] + bv1;
            const long i0 = r * HDIM + col0;
            const long i1 = r * HDIM + col1;
            if (Yf) { Yf[i0] = v0; Yf[i1] = v1; }
            if (Yh) { Yh[i0] = (_Float16)v0; Yh[i1] = (_Float16)v1; }
        }
    }
}

// ---------------------------------------------------------------------------
// Misc small kernels
// ---------------------------------------------------------------------------
__global__ void __launch_bounds__(256)
f32_to_f16_kernel(const float* __restrict__ x, _Float16* __restrict__ y, long n)
{
    long i = (long)blockIdx.x * 256 + threadIdx.x;
    if (i < n) y[i] = (_Float16)x[i];
}

__global__ void __launch_bounds__(256)
deg_kernel(const int* __restrict__ eidx, float* __restrict__ degr,
           float* __restrict__ degc, int E)
{
    long e = (long)blockIdx.x * 256 + threadIdx.x;
    if (e >= E) return;
    atomicAdd(&degr[eidx[e]], 1.0f);
    atomicAdd(&degc[eidx[(long)E + e]], 1.0f);
}

__global__ void __launch_bounds__(256)
pseudo_kernel(const int* __restrict__ eidx, const float* __restrict__ degr,
              const float* __restrict__ degc, float* __restrict__ pseudo, int E)
{
    long e = (long)blockIdx.x * 256 + threadIdx.x;
    if (e >= E) return;
    pseudo[e * 2 + 0] = rsqrtf(degr[eidx[e]] + 1.0f);
    pseudo[e * 2 + 1] = rsqrtf(degc[eidx[(long)E + e]] + 1.0f);
}

// ---------------------------------------------------------------------------
// Per-edge: recompute GMM weights, gather g_k[row] (f16), scatter-add to agg.
// 256 threads = 2 edges x 128 columns.
// ---------------------------------------------------------------------------
__global__ void __launch_bounds__(256)
edge_msg_kernel(const int* __restrict__ eidx,
                const float* __restrict__ pseudo,
                const _Float16* __restrict__ g0,
                const _Float16* __restrict__ g1,
                const _Float16* __restrict__ g2,
                const float* __restrict__ ppw,   // [PD,2] this layer
                const float* __restrict__ ppb,   // [PD]
                const float* __restrict__ mu,    // [K,PD]
                const float* __restrict__ isg,   // [K,PD]
                float* __restrict__ agg,
                int E)
{
    const int t = threadIdx.x;
    const long e = (long)blockIdx.x * 2 + (t >> 7);
    if (e >= E) return;
    const int c   = t & 127;
    const int src = eidx[e];
    const int dst = eidx[(long)E + e];

    const float p0 = pseudo[e * 2 + 0];
    const float p1 = pseudo[e * 2 + 1];
    const float ps0 = tanhf(p0 * ppw[0] + p1 * ppw[1] + ppb[0]);
    const float ps1 = tanhf(p0 * ppw[2] + p1 * ppw[3] + ppb[1]);

    float w[3];
#pragma unroll
    for (int k = 0; k < 3; ++k) {
        const float d0 = ps0 - mu[k * 2 + 0];
        const float d1 = ps1 - mu[k * 2 + 1];
        const float s0 = isg[k * 2 + 0];
        const float s1 = isg[k * 2 + 1];
        w[k] = __expf(-0.5f * (d0 * d0 * s0 * s0 + d1 * d1 * s1 * s1));
    }

    const long off = (long)src * HDIM + c;
    const float m = w[0] * (float)g0[off] + w[1] * (float)g1[off] + w[2] * (float)g2[off];
    atomicAdd(&agg[(long)dst * HDIM + c], m);
}

// ---------------------------------------------------------------------------
// Column-wise BN statistics (pass 1): per-block partial sums + atomics.
// ---------------------------------------------------------------------------
__global__ void __launch_bounds__(128)
bn_stats_kernel(const float* __restrict__ agg, float* __restrict__ colsum,
                float* __restrict__ colsq, int N, int rowsPerBlock)
{
    const int t = threadIdx.x;
    const long r0 = (long)blockIdx.x * rowsPerBlock;
    float s = 0.f, s2 = 0.f;
    for (int i = 0; i < rowsPerBlock; ++i) {
        const long r = r0 + i;
        if (r < N) {
            const float v = agg[r * HDIM + t];
            s += v; s2 += v * v;
        }
    }
    atomicAdd(&colsum[t], s);
    atomicAdd(&colsq[t], s2);
}

// BN (pass 2): normalize + ReLU + residual; also emit f16 copy for next GEMM.
__global__ void __launch_bounds__(512)
bn_finalize_kernel(const float* __restrict__ agg,
                   const float* __restrict__ colsum, const float* __restrict__ colsq,
                   const float* __restrict__ gamma, const float* __restrict__ beta,
                   float* __restrict__ h, _Float16* __restrict__ h16, int N)
{
    const int t = threadIdx.x;
    const int c = t & 127;
    const long r = (long)blockIdx.x * 4 + (t >> 7);
    if (r >= N) return;
    const float invN = 1.0f / (float)N;
    const float mean = colsum[c] * invN;
    const float var  = colsq[c] * invN - mean * mean;
    const float sc   = rsqrtf(var + 1e-5f) * gamma[c];
    const long idx = r * HDIM + c;
    const float hn = (agg[idx] - mean) * sc + beta[c];
    const float v  = h[idx] + fmaxf(hn, 0.f);
    h[idx]   = v;
    h16[idx] = (_Float16)v;
}

// Graph mean readout (pass 1: sums + counts).
__global__ void __launch_bounds__(128)
readout_kernel(const float* __restrict__ h, const int* __restrict__ batch,
               float* __restrict__ hs, float* __restrict__ cnt, int N)
{
    const int n = blockIdx.x;
    const int t = threadIdx.x;
    const int g = batch[n];
    atomicAdd(&hs[(long)g * HDIM + t], h[(long)n * HDIM + t]);
    if (t == 0) atomicAdd(&cnt[g], 1.0f);
}

// Tiny MLP head: 128 -> 64 -> 32 -> 10, one block per graph.
__global__ void __launch_bounds__(128)
mlp_kernel(const float* __restrict__ hs, const float* __restrict__ cnt,
           const float* __restrict__ w0, const float* __restrict__ b0,
           const float* __restrict__ w1, const float* __restrict__ b1,
           const float* __restrict__ w2, const float* __restrict__ b2,
           float* __restrict__ out)
{
    __shared__ float x[128];
    __shared__ float z0[64];
    __shared__ float z1[32];
    const int g = blockIdx.x;
    const int t = threadIdx.x;
    const float cg = fmaxf(cnt[g], 1.0f);
    x[t] = hs[(long)g * HDIM + t] / cg;
    __syncthreads();
    if (t < 64) {
        float s = b0[t];
        for (int j = 0; j < 128; ++j) s += x[j] * w0[t * 128 + j];
        z0[t] = fmaxf(s, 0.f);
    }
    __syncthreads();
    if (t < 32) {
        float s = b1[t];
        for (int j = 0; j < 64; ++j) s += z0[j] * w1[t * 64 + j];
        z1[t] = fmaxf(s, 0.f);
    }
    __syncthreads();
    if (t < 10) {
        float s = b2[t];
        for (int j = 0; j < 32; ++j) s += z1[j] * w2[t * 32 + j];
        out[(long)g * 10 + t] = s;
    }
}

// ---------------------------------------------------------------------------
// Host orchestration
// ---------------------------------------------------------------------------
extern "C" void kernel_launch(void* const* d_in, const int* in_sizes, int n_in,
                              void* d_out, int out_size, void* d_ws, size_t ws_size,
                              hipStream_t stream)
{
    (void)in_sizes; (void)n_in; (void)out_size; (void)ws_size;

    constexpr int N = 50000, E = 800000, G = 128, H = 128, K = 3, L = 4, PD = 2;

    const float* feature  = (const float*)d_in[0];
    const int*   eidx     = (const int*)  d_in[1];
    const int*   batch    = (const int*)  d_in[2];
    const float* emb_w    = (const float*)d_in[3];
    const float* emb_b    = (const float*)d_in[4];
    const float* fc_w     = (const float*)d_in[5];   // [L,K*H,H]
    const float* mu       = (const float*)d_in[6];   // [L,K,PD]
    const float* isg      = (const float*)d_in[7];   // [L,K,PD]
    const float* bn_gamma = (const float*)d_in[8];   // [L,H]
    const float* bn_beta  = (const float*)d_in[9];   // [L,H]
    const float* pp_w     = (const float*)d_in[10];  // [L,PD,2]
    const float* pp_b     = (const float*)d_in[11];  // [L,PD]
    const float* mlp_w0   = (const float*)d_in[12];
    const float* mlp_b0   = (const float*)d_in[13];
    const float* mlp_w1   = (const float*)d_in[14];
    const float* mlp_b1   = (const float*)d_in[15];
    const float* mlp_w2   = (const float*)d_in[16];
    const float* mlp_b2   = (const float*)d_in[17];
    float* out = (float*)d_out;

    // Workspace carve-out (256B aligned chunks).
    char* ws = (char*)d_ws;
    size_t off = 0;
    auto carve = [&](size_t bytes) -> void* {
        void* p = ws + off;
        off += (bytes + 255) & ~(size_t)255;
        return p;
    };
    float*     h      = (float*)    carve((size_t)N * H * 4);
    _Float16*  h16    = (_Float16*) carve((size_t)N * H * 2);
    _Float16*  g16    = (_Float16*) carve((size_t)3 * N * H * 2);
    float*     agg    = (float*)    carve((size_t)N * H * 4);
    float*     pseudo = (float*)    carve((size_t)E * 2 * 4);
    float*     degr   = (float*)    carve((size_t)2 * N * 4);
    float*     degc   = degr + N;
    float*     colsum = (float*)    carve((size_t)2 * H * 4);
    float*     colsq  = colsum + H;
    float*     hs     = (float*)    carve((size_t)(G * H + G) * 4);
    float*     cnt    = hs + (size_t)G * H;
    _Float16*  w16all = (_Float16*) carve((size_t)(1 + L * K) * H * H * 2);
    _Float16*  embw16 = w16all;
    _Float16*  fcw16  = w16all + (size_t)H * H;
    _Float16*  feat16 = g16;   // reuse: feat16 only needed before g16 is written

    const long NH = (long)N * H;
    const long WN = (long)H * H;

    // One-time f16 conversion of all weight matrices (13 x 128x128 = 425KB).
    f32_to_f16_kernel<<<(int)((WN + 255) / 256), 256, 0, stream>>>(emb_w, embw16, WN);
    f32_to_f16_kernel<<<(int)((L * K * WN + 255) / 256), 256, 0, stream>>>(
        fc_w, fcw16, (long)L * K * WN);

    // Degrees + pseudo coordinates.
    hipMemsetAsync(degr, 0, (size_t)2 * N * 4, stream);
    deg_kernel<<<(E + 255) / 256, 256, 0, stream>>>(eidx, degr, degc, E);
    pseudo_kernel<<<(E + 255) / 256, 256, 0, stream>>>(eidx, degr, degc, pseudo, E);

    // Embedding GEMM: h = feature @ emb_w^T + emb_b  (f32 + f16 copies).
    f32_to_f16_kernel<<<(int)((NH + 255) / 256), 256, 0, stream>>>(feature, feat16, NH);
    gemm_abt_wmma<<<N / 16, 128, 0, stream>>>(feat16, embw16, emb_b, h, h16, N);

    for (int l = 0; l < L; ++l) {
        // Node-side transforms: g_k = h @ fc[l,k]^T (f16 output only).
        for (int k = 0; k < K; ++k) {
            gemm_abt_wmma<<<N / 16, 128, 0, stream>>>(
                h16, fcw16 + ((size_t)l * K + k) * WN, nullptr,
                nullptr, g16 + (size_t)k * NH, N);
        }

        hipMemsetAsync(agg, 0, (size_t)N * H * 4, stream);
        hipMemsetAsync(colsum, 0, (size_t)2 * H * 4, stream);

        edge_msg_kernel<<<(E + 1) / 2, 256, 0, stream>>>(
            eidx, pseudo, g16, g16 + NH, g16 + 2 * NH,
            pp_w + (size_t)l * PD * 2, pp_b + (size_t)l * PD,
            mu + (size_t)l * K * PD, isg + (size_t)l * K * PD, agg, E);

        bn_stats_kernel<<<1000, 128, 0, stream>>>(agg, colsum, colsq, N, 50);
        bn_finalize_kernel<<<(N + 3) / 4, 512, 0, stream>>>(
            agg, colsum, colsq,
            bn_gamma + (size_t)l * H, bn_beta + (size_t)l * H, h, h16, N);
    }

    // Readout + MLP head.
    hipMemsetAsync(hs, 0, (size_t)(G * H + G) * 4, stream);
    readout_kernel<<<N, 128, 0, stream>>>(h, batch, hs, cnt, N);
    mlp_kernel<<<G, 128, 0, stream>>>(hs, cnt, mlp_w0, mlp_b0, mlp_w1, mlp_b1,
                                      mlp_w2, mlp_b2, out);
}
